// GPT_14319420965176
// MI455X (gfx1250) — matrix operand order repaired and checked by reading, not compiled, not run on hardware
//
#include <hip/hip_runtime.h>
#include <math.h>

// ---------------------------------------------------------------------------
// GPT-2 small forward pass for MI455X (gfx1250, wave32, WMMA bf16 16x16x32)
// ---------------------------------------------------------------------------
typedef __bf16 bf16;
typedef __attribute__((ext_vector_type(16))) __bf16 v16bf;
typedef __attribute__((ext_vector_type(8)))  __bf16 bf16x8;
typedef __attribute__((ext_vector_type(8)))  float  v8f;

#define EMB   768
#define NHEAD 12
#define HDIM  64
#define SEQ   1024
#define NB    4
#define BT    (NB * SEQ)          // 4096 rows (always a multiple of 128)
#define NLAY  12
#define NVOC  50257

__device__ __forceinline__ v8f wmma_bf16(v16bf a, v16bf b, v8f c) {
  // D = A(16x32 bf16) x B(32x16 bf16) + C(16x16 f32)
  return __builtin_amdgcn_wmma_f32_16x16x32_bf16(false, a, false, b,
                                                 (short)0, c, false, false);
}

// Assemble a 16-element fragment from two 8-element (16B) vector loads.
__device__ __forceinline__ v16bf ld_frag2(const bf16* p0, const bf16* p1) {
  bf16x8 lo = *(const bf16x8*)p0;
  bf16x8 hi = *(const bf16x8*)p1;
  return __builtin_shufflevector(lo, hi, 0, 1, 2, 3, 4, 5, 6, 7,
                                 8, 9, 10, 11, 12, 13, 14, 15);
}

// 16-lane rotate-reductions in pure VALU via DPP16 row_ror (no ds_bpermute).
template <int CTRL>
__device__ __forceinline__ float dpp_movf(float x) {
  return __int_as_float(__builtin_amdgcn_update_dpp(
      0, __float_as_int(x), CTRL, 0xf, 0xf, true));
}
__device__ __forceinline__ float red16_max(float x) {
  x = fmaxf(x, dpp_movf<0x121>(x));   // row_ror:1
  x = fmaxf(x, dpp_movf<0x122>(x));   // row_ror:2
  x = fmaxf(x, dpp_movf<0x124>(x));   // row_ror:4
  x = fmaxf(x, dpp_movf<0x128>(x));   // row_ror:8
  return x;
}
__device__ __forceinline__ float red16_sum(float x) {
  x += dpp_movf<0x121>(x);
  x += dpp_movf<0x122>(x);
  x += dpp_movf<0x124>(x);
  x += dpp_movf<0x128>(x);
  return x;
}

// ---------------------------------------------------------------------------
// fp32 -> bf16 conversion (weights converted once per launch)
// ---------------------------------------------------------------------------
__global__ void cvt_f32_to_bf16(const float* __restrict__ src,
                                bf16* __restrict__ dst, long n) {
  long i = (long)blockIdx.x * blockDim.x + threadIdx.x;
  long stride = (long)gridDim.x * blockDim.x;
  for (; i < n; i += stride) dst[i] = (bf16)src[i];
}

// ---------------------------------------------------------------------------
// Embedding: x[b,t,:] = wte[idx[b,t],:] + wpe[t,:]
// ---------------------------------------------------------------------------
__global__ void embed_kernel(const int* __restrict__ idx,
                             const float* __restrict__ wte,
                             const float* __restrict__ wpe,
                             float* __restrict__ x) {
  int row = blockIdx.x;               // 0..BT-1
  int t   = row % SEQ;
  long tok = idx[row];
  const float* we = wte + tok * (long)EMB;
  const float* pe = wpe + (long)t * EMB;
  float* xr = x + (long)row * EMB;
  for (int e = threadIdx.x; e < EMB; e += blockDim.x)
    xr[e] = we[e] + pe[e];
}

// ---------------------------------------------------------------------------
// LayerNorm: fp32 in -> bf16 out (input to the next WMMA GEMM)
// ---------------------------------------------------------------------------
__global__ __launch_bounds__(256)
void ln_kernel(const float* __restrict__ x, const float* __restrict__ g,
               const float* __restrict__ b, bf16* __restrict__ out) {
  __shared__ float red[256];
  int row = blockIdx.x;
  const float* xr = x + (long)row * EMB;
  float s = 0.f, s2 = 0.f;
  for (int e = threadIdx.x; e < EMB; e += 256) {
    float v = xr[e];
    s += v; s2 += v * v;
  }
  red[threadIdx.x] = s; __syncthreads();
  for (int o = 128; o > 0; o >>= 1) {
    if (threadIdx.x < o) red[threadIdx.x] += red[threadIdx.x + o];
    __syncthreads();
  }
  float mean = red[0] * (1.0f / EMB);
  __syncthreads();
  red[threadIdx.x] = s2; __syncthreads();
  for (int o = 128; o > 0; o >>= 1) {
    if (threadIdx.x < o) red[threadIdx.x] += red[threadIdx.x + o];
    __syncthreads();
  }
  float var = red[0] * (1.0f / EMB) - mean * mean;
  float rstd = rsqrtf(var + 1e-5f);
  bf16* orow = out + (long)row * EMB;
  for (int e = threadIdx.x; e < EMB; e += 256)
    orow[e] = (bf16)((xr[e] - mean) * rstd * g[e] + b[e]);
}

// ---------------------------------------------------------------------------
// Tiled bf16 WMMA GEMM: C[M,N] = A[M,K] @ B[K,N]
//   block tile 128(M) x 128(N) x 32(K), 256 threads = 8 wave32 waves,
//   each wave computes a 32x64 micro-tile (2x4 v_wmma_f32_16x16x32_bf16).
//   Double-buffered LDS: next K-tile is loaded into buf^1 while computing
//   from buf -> one s_barrier per iteration, HBM latency hidden under WMMA.
//   B is staged in LDS pre-transposed (Bt[n][k]) so every fragment read is
//   two contiguous ds_load_b128.
//   Invariants: M % 128 == 0; TRANSB==false => N % 128 == 0 (branchless);
//   TRANSB==true => B stored [N,K] (lm_head, N = 50257, guards kept).
// ---------------------------------------------------------------------------
template <bool TRANSB>
__global__ __launch_bounds__(256)
void gemm_bf16_kernel(const bf16* __restrict__ A, const bf16* __restrict__ Bm,
                      const float* __restrict__ bias,
                      const float* __restrict__ resid,
                      float* __restrict__ outf, bf16* __restrict__ outb,
                      int M, int N, int K, int gelu) {
  __shared__ __align__(16) bf16 As[2][128][40];   // [m][k], padded stride
  __shared__ __align__(16) bf16 Bt[2][128][40];   // [n][k], padded stride

  const int tid  = threadIdx.x;
  const int lane = tid & 31;
  const int wave = tid >> 5;
  const int l    = lane & 15;
  const int grp  = lane >> 4;          // 0 or 1
  const int wm   = wave >> 1;          // 0..3 -> M sub-tile (32 rows)
  const int wn   = wave & 1;           // 0..1 -> N sub-tile (64 cols)
  const int bm0  = blockIdx.y * 128;
  const int bn0  = blockIdx.x * 128;

  auto load_tile = [&](int b, int kk) {
    // A tile (128x32): 512 strips of 8 bf16, 2 per thread (branchless)
#pragma unroll
    for (int s = 0; s < 2; ++s) {
      int strip = tid * 2 + s;
      int am = strip >> 2;
      int ak = (strip & 3) * 8;
      const bf16* src = A + (long)(bm0 + am) * K + kk + ak;
      bf16x8 v = *(const bf16x8*)src;
      __builtin_prefetch(src + 32, 0, 3);
      *(bf16x8*)&As[b][am][ak] = v;
    }
    if (!TRANSB) {
      // B tile (32x128) staged transposed into Bt[n][k]
#pragma unroll
      for (int s = 0; s < 2; ++s) {
        int strip = tid * 2 + s;         // 512 strips of 8 along n
        int bk = strip >> 4;
        int bn = (strip & 15) * 8;
        const bf16* src = Bm + (long)(kk + bk) * N + bn0 + bn;
        bf16x8 v = *(const bf16x8*)src;
        __builtin_prefetch(src + 32L * N, 0, 3);
#pragma unroll
        for (int j = 0; j < 8; ++j) Bt[b][bn + j][bk] = v[j];
      }
    } else {
#pragma unroll
      for (int s = 0; s < 2; ++s) {
        int strip = tid * 2 + s;         // 512 strips of 8 along k
        int n  = strip >> 2;
        int ks = (strip & 3) * 8;
        bf16x8 v = {};
        if (bn0 + n < N) {
          const bf16* src = Bm + (long)(bn0 + n) * K + kk + ks;
          v = *(const bf16x8*)src;
          __builtin_prefetch(src + 32, 0, 3);
        }
        *(bf16x8*)&Bt[b][n][ks] = v;     // straight 16B copy
      }
    }
  };

  v8f acc[2][4] = {};

  load_tile(0, 0);
  __syncthreads();
  int buf = 0;

  for (int k0 = 0; k0 < K; k0 += 32) {
    if (k0 + 32 < K) load_tile(buf ^ 1, k0 + 32);   // prefetch next tile

    // ---- fragments: all reads are contiguous 16B LDS vector loads
    v16bf afr[2], bfr[4];
#pragma unroll
    for (int i = 0; i < 2; ++i) {
      const bf16* ap = &As[buf][wm * 32 + i * 16 + l][0];
      afr[i] = *(const v16bf*)(ap + grp * 16);
    }
#pragma unroll
    for (int i = 0; i < 4; ++i)
      bfr[i] = *(const v16bf*)&Bt[buf][wn * 64 + i * 16 + l][grp * 16];

#pragma unroll
    for (int mi = 0; mi < 2; ++mi)
#pragma unroll
      for (int ni = 0; ni < 4; ++ni)
        acc[mi][ni] = wmma_bf16(afr[mi], bfr[ni], acc[mi][ni]);

    __syncthreads();     // next tile fully staged + this tile's reads done
    buf ^= 1;
  }

  // ---- epilogue (N guard only needed for the transB / lm_head path)
#pragma unroll
  for (int mi = 0; mi < 2; ++mi) {
#pragma unroll
    for (int ni = 0; ni < 4; ++ni) {
#pragma unroll
      for (int i = 0; i < 8; ++i) {
        int m = bm0 + wm * 32 + mi * 16 + grp * 8 + i;
        int n = bn0 + wn * 64 + ni * 16 + l;
        if (!TRANSB || n < N) {
          float v = acc[mi][ni][i];
          if (bias)  v += bias[n];
          if (resid) v += resid[(long)m * N + n];
          if (gelu) {
            float u = v + 0.044715f * v * v * v;
            v = 0.5f * v * (1.0f + tanhf(0.7978845608028654f * u));
          }
          if (outf) outf[(long)m * N + n] = v;
          if (outb) outb[(long)m * N + n] = (bf16)v;
        }
      }
    }
  }
}

// ---------------------------------------------------------------------------
// Causal flash attention, fully WMMA-based, double-buffered K/V staging.
//   qkv: [BT, 3E] bf16 (q | k | v per row).  y: [BT, E] bf16.
//   One block = one (batch,head) x 128 query rows; each wave owns 16 queries.
//   Chunk loop is block-uniform: next 32-key K/V chunk staged into buf^1
//   while computing from buf (one barrier per chunk). Waves beyond their
//   causal range skip compute but keep the barriers.
//   S = Q@K^T (4 WMMAs), online softmax (DPP16 row_ror reductions),
//   O += P@V (4 WMMAs) per chunk. P C-frag -> A-frag transpose goes through
//   per-wave LDS guarded by the wave-local `s_wait_dscnt 0`.
// ---------------------------------------------------------------------------
__global__ __launch_bounds__(256)
void attn_kernel(const bf16* __restrict__ qkv, bf16* __restrict__ y) {
  __shared__ __align__(16) bf16 Ks[2][32][72];    // K chunk  [key][d]
  __shared__ __align__(16) bf16 Vt[2][64][40];    // V chunk  [d][key]
  __shared__ __align__(16) bf16 Pt[8][16][40];    // per-wave P tile (16x32)

  const int tid  = threadIdx.x;
  const int lane = tid & 31;
  const int wave = tid >> 5;
  const int l    = lane & 15;
  const int grp  = lane >> 4;
  const int bh = blockIdx.x;
  const int bb = bh / NHEAD;
  const int hh = bh % NHEAD;
  const int q0 = blockIdx.y * 128 + wave * 16;

  const long rs = 3 * EMB;
  const bf16* qbase = qkv + (long)bb * SEQ * rs + hh * HDIM;
  const bf16* kbase = qbase + EMB;
  const bf16* vbase = qbase + 2 * EMB;

  const int kload_row = tid >> 3;          // 0..31
  const int kload_d   = (tid & 7) * 8;     // 0..56

  auto load_chunk = [&](int b, int kb) {
    const long grow = (long)(kb + kload_row) * rs + kload_d;
    bf16x8 kv = *(const bf16x8*)(kbase + grow);
    bf16x8 vv = *(const bf16x8*)(vbase + grow);
    *(bf16x8*)&Ks[b][kload_row][kload_d] = kv;
#pragma unroll
    for (int j = 0; j < 8; ++j) Vt[b][kload_d + j][kload_row] = vv[j];
  };

  // Q fragments (16 rows x 64 head dims = two 16x32 A-frags), contiguous loads
  v16bf qf[2];
#pragma unroll
  for (int f = 0; f < 2; ++f)
    qf[f] = *(const v16bf*)(qbase + (long)(q0 + l) * rs + f * 32 + grp * 16);

  v8f O[4] = {};                        // 16x64 output accumulator (4 C-frags)
  float mrow[8], lrow[8];
#pragma unroll
  for (int i = 0; i < 8; ++i) { mrow[i] = -1e30f; lrow[i] = 0.f; }

  const int nchunk = blockIdx.y * 4 + 4;   // chunks covering keys 0..q_max

  load_chunk(0, 0);
  __syncthreads();
  int buf = 0;

  for (int c = 0; c < nchunk; ++c) {
    const int kb = c * 32;
    if (c + 1 < nchunk) load_chunk(buf ^ 1, kb + 32);   // stage next chunk

    if (kb <= q0 + 15) {                 // uniform per wave (causal skip)
      // S[s] covers keys kb + s*16 .. kb + s*16 + 15
      v8f S[2] = {};
#pragma unroll
      for (int s = 0; s < 2; ++s) {
#pragma unroll
        for (int f = 0; f < 2; ++f) {
          v16bf bv = *(const v16bf*)&Ks[buf][s * 16 + l][f * 32 + grp * 16];
          S[s] = wmma_bf16(qf[f], bv, S[s]);
        }
      }

      // online softmax over the 32-key chunk (DPP16 row reductions, no DS)
#pragma unroll
      for (int i = 0; i < 8; ++i) {
        int qg = q0 + grp * 8 + i;
        float s0 = S[0][i] * 0.125f;     // 1/sqrt(64)
        float s1 = S[1][i] * 0.125f;
        if (kb + l      > qg) s0 = -1e30f;
        if (kb + 16 + l > qg) s1 = -1e30f;
        float mnew = fmaxf(mrow[i], red16_max(fmaxf(s0, s1)));
        float e0 = __expf(s0 - mnew);
        float e1 = __expf(s1 - mnew);
        float rsum = red16_sum(e0 + e1);
        float scale = __expf(mrow[i] - mnew);
        lrow[i] = lrow[i] * scale + rsum;
        mrow[i] = mnew;
#pragma unroll
        for (int t = 0; t < 4; ++t) O[t][i] *= scale;
        Pt[wave][grp * 8 + i][l]      = (bf16)e0;
        Pt[wave][grp * 8 + i][16 + l] = (bf16)e1;
      }

      // wave-local LDS RAW fence (per-wave Pt region; no block barrier)
      asm volatile("s_wait_dscnt 0" ::: "memory");

      // reload P as A-frag (16 queries x 32 keys), contiguous loads
      const bf16* pp = &Pt[wave][l][0];
      v16bf pf = *(const v16bf*)(pp + grp * 16);

      // O += P @ V  (reduction over 32 keys, 4 d-tiles of 16)
#pragma unroll
      for (int t = 0; t < 4; ++t) {
        v16bf vv = *(const v16bf*)&Vt[buf][t * 16 + l][grp * 16];
        O[t] = wmma_bf16(pf, vv, O[t]);
      }
    }
    __syncthreads();     // next chunk staged + this chunk's reads done
    buf ^= 1;
  }

  // normalize and store y (bf16, feeds the proj GEMM)
#pragma unroll
  for (int i = 0; i < 8; ++i) {
    float inv = 1.0f / lrow[i];
    int r = q0 + grp * 8 + i;
    bf16* yr = y + (long)(bb * SEQ + r) * EMB + hh * HDIM;
#pragma unroll
    for (int t = 0; t < 4; ++t) yr[t * 16 + l] = (bf16)(O[t][i] * inv);
  }
}

// ---------------------------------------------------------------------------
// Host-side orchestration
// ---------------------------------------------------------------------------
extern "C" void kernel_launch(void* const* d_in, const int* in_sizes, int n_in,
                              void* d_out, int out_size, void* d_ws, size_t ws_size,
                              hipStream_t stream) {
  (void)in_sizes; (void)n_in; (void)out_size; (void)ws_size;

  const int*   idx    = (const int*)  d_in[0];
  const float* wte    = (const float*)d_in[1];
  const float* wpe    = (const float*)d_in[2];
  const float* ln1g   = (const float*)d_in[3];
  const float* ln1b   = (const float*)d_in[4];
  const float* attnw  = (const float*)d_in[5];
  const float* attnb  = (const float*)d_in[6];
  const float* projw  = (const float*)d_in[7];
  const float* projb  = (const float*)d_in[8];
  const float* ln2g   = (const float*)d_in[9];
  const float* ln2b   = (const float*)d_in[10];
  const float* fcw    = (const float*)d_in[11];
  const float* fcb    = (const float*)d_in[12];
  const float* mprojw = (const float*)d_in[13];
  const float* mprojb = (const float*)d_in[14];
  const float* lnfg   = (const float*)d_in[15];
  const float* lnfb   = (const float*)d_in[16];
  float* logits = (float*)d_out;

  // workspace carve-up (256B aligned)
  char* p = (char*)d_ws;
  auto take = [&](size_t bytes) -> char* {
    char* r = p;
    p += (bytes + 255) & ~(size_t)255;
    return r;
  };
  bf16* wteB   = (bf16*)take((size_t)NVOC * EMB * 2);
  bf16* wqkvB  = (bf16*)take((size_t)NLAY * EMB * 3 * EMB * 2);
  bf16* wprojB = (bf16*)take((size_t)NLAY * EMB * EMB * 2);
  bf16* wfcB   = (bf16*)take((size_t)NLAY * EMB * 4 * EMB * 2);
  bf16* wmprB  = (bf16*)take((size_t)NLAY * 4 * EMB * EMB * 2);
  float* xbuf  = (float*)take((size_t)BT * EMB * 4);
  bf16* hbuf   = (bf16*) take((size_t)BT * EMB * 2);
  bf16* qkvbuf = (bf16*) take((size_t)BT * 3 * EMB * 2);
  bf16* ybuf   = (bf16*) take((size_t)BT * EMB * 2);
  bf16* hfcbuf = (bf16*) take((size_t)BT * 4 * EMB * 2);

  // 1) one-time weight down-conversion to bf16
  cvt_f32_to_bf16<<<2048, 256, 0, stream>>>(wte,    wteB,   (long)NVOC * EMB);
  cvt_f32_to_bf16<<<2048, 256, 0, stream>>>(attnw,  wqkvB,  (long)NLAY * EMB * 3 * EMB);
  cvt_f32_to_bf16<<<2048, 256, 0, stream>>>(projw,  wprojB, (long)NLAY * EMB * EMB);
  cvt_f32_to_bf16<<<2048, 256, 0, stream>>>(fcw,    wfcB,   (long)NLAY * EMB * 4 * EMB);
  cvt_f32_to_bf16<<<2048, 256, 0, stream>>>(mprojw, wmprB,  (long)NLAY * 4 * EMB * EMB);

  // 2) embeddings
  embed_kernel<<<BT, 256, 0, stream>>>(idx, wte, wpe, xbuf);

  const dim3 blk(256);
  auto gemm_grid = [](int M, int N) { return dim3((N + 127) / 128, (M + 127) / 128); };

  // 3) transformer layers
  for (int layer = 0; layer < NLAY; ++layer) {
    const bf16* Wqkv = wqkvB  + (long)layer * EMB * 3 * EMB;
    const bf16* Wprj = wprojB + (long)layer * EMB * EMB;
    const bf16* Wfc  = wfcB   + (long)layer * EMB * 4 * EMB;
    const bf16* Wmp  = wmprB  + (long)layer * 4 * EMB * EMB;

    // LN1 -> h
    ln_kernel<<<BT, blk, 0, stream>>>(xbuf, ln1g + (long)layer * EMB,
                                      ln1b + (long)layer * EMB, hbuf);
    // qkv = h @ Wqkv + b   [BT, 3E] bf16
    gemm_bf16_kernel<false><<<gemm_grid(BT, 3 * EMB), blk, 0, stream>>>(
        hbuf, Wqkv, attnb + (long)layer * 3 * EMB,
        nullptr, nullptr, qkvbuf, BT, 3 * EMB, EMB, 0);
    // causal flash attention -> y (bf16)
    attn_kernel<<<dim3(NB * NHEAD, SEQ / 128), blk, 0, stream>>>(qkvbuf, ybuf);
    // x = x + y @ Wproj + b   (fp32 residual stream)
    gemm_bf16_kernel<false><<<gemm_grid(BT, EMB), blk, 0, stream>>>(
        ybuf, Wprj, projb + (long)layer * EMB,
        xbuf, xbuf, nullptr, BT, EMB, EMB, 0);
    // LN2 -> h
    ln_kernel<<<BT, blk, 0, stream>>>(xbuf, ln2g + (long)layer * EMB,
                                      ln2b + (long)layer * EMB, hbuf);
    // hfc = gelu(h @ Wfc + b)   [BT, 4E] bf16
    gemm_bf16_kernel<false><<<gemm_grid(BT, 4 * EMB), blk, 0, stream>>>(
        hbuf, Wfc, fcb + (long)layer * 4 * EMB,
        nullptr, nullptr, hfcbuf, BT, 4 * EMB, EMB, 1);
    // x = x + hfc @ Wmproj + b
    gemm_bf16_kernel<false><<<gemm_grid(BT, EMB), blk, 0, stream>>>(
        hfcbuf, Wmp, mprojb + (long)layer * EMB,
        xbuf, xbuf, nullptr, BT, EMB, 4 * EMB, 0);
  }

  // 4) final LN -> h
  ln_kernel<<<BT, blk, 0, stream>>>(xbuf, lnfg, lnfb, hbuf);

  // 5) logits = h @ wte^T  (transB path, fp32 out, N = 50257 with guards)
  gemm_bf16_kernel<true><<<gemm_grid(BT, NVOC), blk, 0, stream>>>(
      hbuf, wteB, nullptr, nullptr, logits, nullptr, BT, NVOC, EMB, 0);
}
